// Attention_43293270343986
// MI455X (gfx1250) — matrix operand order repaired and checked by reading, not compiled, not run on hardware
//
#include <hip/hip_runtime.h>

typedef __attribute__((ext_vector_type(16))) __bf16 v16bf;
typedef __attribute__((ext_vector_type(8)))  float  v8f;
typedef __attribute__((ext_vector_type(4)))  unsigned int v4u;
typedef __attribute__((ext_vector_type(8)))  int    v8i;
typedef __attribute__((ext_vector_type(4)))  int    v4i;

#if defined(__has_builtin)
#if __has_builtin(__builtin_amdgcn_tensor_load_to_lds) && \
    __has_builtin(__builtin_amdgcn_s_wait_tensorcnt)
#define HAVE_TDM 1
#endif
#endif
#ifndef HAVE_TDM
#define HAVE_TDM 0
#endif

// ---------------------------------------------------------------------------
// Fragment loaders per CDNA5 ISA 7.12.2 (wave32, 16-bit WMMA operands).
// A (16x32, MxK): lane (hl*16+r) holds row r; elem e -> K = (e<8?0:16)+hl*8+(e&7)
// B (32x16, KxN): lane (hl*16+c) holds col c; elem e -> K = (e<8?0:16)+hl*8+(e&7)
// C/D (16x16 f32): vgpr j, lane (hl*16+c) -> row j + 8*hl, col c
// ---------------------------------------------------------------------------
__device__ inline v16bf load_a_frag(const __bf16* base, int row0, int lane,
                                    int ld, int k0) {
  const int hl = lane >> 4, r = lane & 15;
  union { v16bf v; uint4 u[2]; } f;
  const __bf16* p = base + (size_t)(row0 + r) * ld + k0 + hl * 8;
  f.u[0] = *(const uint4*)(p);
  f.u[1] = *(const uint4*)(p + 16);
  return f.v;
}

__device__ inline v16bf load_b_frag(const __bf16* base, int col0, int lane,
                                    int ld, int k0) {
  const int hl = lane >> 4, c = lane & 15;
  union { v16bf v; uint4 u[2]; } f;
  const __bf16* p = base + (size_t)(col0 + c) * ld + k0 + hl * 8;
  f.u[0] = *(const uint4*)(p);
  f.u[1] = *(const uint4*)(p + 16);
  return f.v;
}

#if HAVE_TDM
// ---------------------------------------------------------------------------
// TDM: async-DMA a 2D tile (tile_d1 rows x tile_d0 elements, bf16) from a
// row-major tensor (row stride = stride0 elements) into LDS, packed row-major.
// D# per CDNA5 ISA ch.8: group0 {count=1, lds_addr, global_addr[56:0], type=2},
// group1 {data_size=1(2B), tensor dims, tile dims, dim0 stride}. 2D -> groups
// 2/3 zero. Issued by one wave; completion via TENSORcnt.
// This toolchain's builtin takes 6 args (g0 v4u, g1 v8i, g2 v4i, g3 v4i,
// extra v8i, cpol i32).
// ---------------------------------------------------------------------------
__device__ inline void tdm_load_2d(const __bf16* gptr, unsigned lds_off,
                                   unsigned tile_d0, unsigned tile_d1,
                                   unsigned tensor_d0, unsigned tensor_d1,
                                   unsigned stride0) {
  unsigned long long ga = (unsigned long long)(const void*)gptr;
  v4u g0;
  g0.x = 1u;                                       // count=1 (valid descriptor)
  g0.y = lds_off;                                  // lds_addr (bytes)
  g0.z = (unsigned)(ga & 0xffffffffu);             // global_addr[31:0]
  g0.w = (unsigned)((ga >> 32) & 0x01ffffffu) | (2u << 30);  // [56:32] | type=2
  v8i g1;
  g1[0] = (int)(1u << 16);                         // wg_mask=0, data_size=2B
  g1[1] = (int)((tensor_d0 & 0xffffu) << 16);      // tensor_dim0[15:0]
  g1[2] = (int)((tensor_d0 >> 16) | ((tensor_d1 & 0xffffu) << 16));
  g1[3] = (int)((tensor_d1 >> 16) | (tile_d0 << 16));  // tile_dim0
  g1[4] = (int)(tile_d1 & 0xffffu);                // tile_dim1, tile_dim2=0
  g1[5] = (int)stride0;                            // tensor_dim0_stride[31:0]
  g1[6] = 0;                                       // stride0 hi, stride1 lo
  g1[7] = 0;
  v4i gz4 = {0, 0, 0, 0};
  v8i gz8 = {0, 0, 0, 0, 0, 0, 0, 0};
  __builtin_amdgcn_tensor_load_to_lds(g0, g1, gz4, gz4, gz8, 0);
}
#endif

__device__ inline unsigned lds_offset_of(const void* p) {
  return (unsigned)(size_t)p;  // LDS aperture: addr[31:0] is the LDS offset
}

// ---------------------------------------------------------------------------
__global__ void cvt_bf16(const float* __restrict__ in, __bf16* __restrict__ out,
                         size_t n) {
  size_t i = (size_t)blockIdx.x * blockDim.x + threadIdx.x;
  size_t stride = (size_t)gridDim.x * blockDim.x;
  for (; i < n; i += stride) out[i] = (__bf16)in[i];
}

// ---------------------------------------------------------------------------
// C = A(MxK,bf16) * B(KxN,bf16), f32 accum. 128 threads (4 waves, 2x2),
// tile M=64 x N=128, K-step 64. A tile DMA'd by TDM (row-major [m][k], ld 64);
// B tile manually transposed into Bs[n][k] while the TDM runs.
// ---------------------------------------------------------------------------
template <typename OutT>
__global__ __launch_bounds__(128)
void gemm_bf16(const __bf16* __restrict__ A, const __bf16* __restrict__ B,
               OutT* __restrict__ C, int M, int N, int K) {
  __shared__ __bf16 As[64 * 64];
  __shared__ __bf16 Bs[128 * 64];
  const int tid = threadIdx.x;
  const int lane = tid & 31;
  const int wave = tid >> 5;
  const int wy = wave >> 1, wx = wave & 1;
  const int mBase = blockIdx.y * 64;
  const int nBase = blockIdx.x * 128;

  const v8f vzero = {0.f, 0.f, 0.f, 0.f, 0.f, 0.f, 0.f, 0.f};
  v8f acc[2][4];
#pragma unroll
  for (int i = 0; i < 2; ++i)
#pragma unroll
    for (int j = 0; j < 4; ++j) acc[i][j] = vzero;

  for (int k0 = 0; k0 < K; k0 += 64) {
#if HAVE_TDM
    if (wave == 0) {  // async DMA of 64x64 A tile while B is staged below
      tdm_load_2d(A + (size_t)mBase * K + k0, lds_offset_of(As),
                  64u, 64u, (unsigned)K, (unsigned)M, (unsigned)K);
    }
#else
    {
      int r = tid >> 1, cp = (tid & 1) * 32;
      const uint4* g = (const uint4*)(A + (size_t)(mBase + r) * K + k0 + cp);
      uint4* s = (uint4*)(As + r * 64 + cp);
      s[0] = g[0]; s[1] = g[1]; s[2] = g[2]; s[3] = g[3];
    }
#endif
    {  // B tile, transposed scatter into Bs[n][k]
      int u = tid & 63;      // covers cols 2u, 2u+1
      int krow = tid >> 6;   // 0..1
#pragma unroll
      for (int kk = 0; kk < 64; kk += 2) {
        int k = kk + krow;
        union { unsigned w; __bf16 e[2]; } t;
        t.w = *(const unsigned*)(B + (size_t)(k0 + k) * N + nBase + u * 2);
        Bs[(u * 2) * 64 + k] = t.e[0];
        Bs[(u * 2 + 1) * 64 + k] = t.e[1];
      }
    }
#if HAVE_TDM
    if (wave == 0) __builtin_amdgcn_s_wait_tensorcnt(0);
#endif
    __syncthreads();
#pragma unroll
    for (int kc = 0; kc < 64; kc += 32) {
      v16bf a0 = load_a_frag(As, wy * 32, lane, 64, kc);
      v16bf a1 = load_a_frag(As, wy * 32 + 16, lane, 64, kc);
#pragma unroll
      for (int n = 0; n < 4; ++n) {
        v16bf b = load_b_frag(Bs, wx * 64 + n * 16, lane, 64, kc);
        acc[0][n] = __builtin_amdgcn_wmma_f32_16x16x32_bf16(
            false, a0, false, b, (short)0, acc[0][n], false, false);
        acc[1][n] = __builtin_amdgcn_wmma_f32_16x16x32_bf16(
            false, a1, false, b, (short)0, acc[1][n], false, false);
      }
    }
    __syncthreads();
  }

  const int hl = lane >> 4, c = lane & 15;
#pragma unroll
  for (int mi = 0; mi < 2; ++mi)
#pragma unroll
    for (int ni = 0; ni < 4; ++ni)
#pragma unroll
      for (int j = 0; j < 8; ++j) {
        int row = mBase + wy * 32 + mi * 16 + j + 8 * hl;
        int col = nBase + wx * 64 + ni * 16 + c;
        C[(size_t)row * N + col] = (OutT)acc[mi][ni][j];
      }
}

// ---------------------------------------------------------------------------
__global__ void rope_kernel(__bf16* __restrict__ X, int H, int total) {
  int idx = blockIdx.x * blockDim.x + threadIdx.x;
  if (idx >= total) return;
  int d = idx & 31;
  int rest = idx >> 5;                 // (b*S + s)*H + h
  int s = (rest / H) % 2048;
  float inv = __powf(10000.0f, -(float)d / 32.0f);
  float ang = (float)s * inv;
  float cs = cosf(ang), sn = sinf(ang);
  size_t base = (size_t)rest * 64;
  float x0 = (float)X[base + d];
  float x1 = (float)X[base + d + 32];
  X[base + d] = (__bf16)(x0 * cs - x1 * sn);
  X[base + d + 32] = (__bf16)(x1 * cs + x0 * sn);
}

// ---------------------------------------------------------------------------
// Causal GQA flash attention. grid = (S/64, HQ, B); block = 128 (4 waves);
// wave handles 16 query rows. Q and K tiles are TDM DMAs; V is manually
// transposed into [d][key] so PV B-fragments are contiguous.
// ---------------------------------------------------------------------------
__global__ __launch_bounds__(128)
void flash_attn(const __bf16* __restrict__ Q, const __bf16* __restrict__ Kt,
                const __bf16* __restrict__ Vt, __bf16* __restrict__ O) {
  const int S = 2048, HQ = 32, HKV = 8, DH = 64;
  __shared__ __bf16 Qs[64 * 64];      // [row][d]
  __shared__ __bf16 Ks[32 * 64];      // [key][d]
  __shared__ __bf16 Vs[64 * 32];      // [d][key]
  __shared__ __bf16 Ps[4][16 * 32];   // per-wave probs staging

  const int tid = threadIdx.x, lane = tid & 31, wave = tid >> 5;
  const int qb = blockIdx.x, h = blockIdx.y, b = blockIdx.z;
  const int hk = h >> 2;              // n_rep = 4
  const int q0 = qb * 64;
  const int hl = lane >> 4, c = lane & 15;
  const int wm = wave * 16;

#if HAVE_TDM
  if (wave == 0) {  // 64x64 Q tile, row stride HQ*DH
    tdm_load_2d(Q + (((size_t)(b * S + q0) * HQ) + h) * DH, lds_offset_of(Qs),
                64u, 64u, 64u, (unsigned)(64 * HQ), (unsigned)(HQ * DH));
  }
#else
  {
    int r = tid >> 1, dp = (tid & 1) * 32;
    const uint4* g =
        (const uint4*)(Q + (((size_t)(b * S + q0 + r) * HQ) + h) * DH + dp);
    uint4* s = (uint4*)(Qs + r * 64 + dp);
    s[0] = g[0]; s[1] = g[1]; s[2] = g[2]; s[3] = g[3];
  }
#endif

  float m_st[8], l_st[8], alpha[8], p0[8], p1[8];
  const v8f vzero = {0.f, 0.f, 0.f, 0.f, 0.f, 0.f, 0.f, 0.f};
  v8f acc[4];
#pragma unroll
  for (int j = 0; j < 8; ++j) { m_st[j] = -__builtin_inff(); l_st[j] = 0.f; }
#pragma unroll
  for (int n = 0; n < 4; ++n) acc[n] = vzero;

  for (int k0 = 0; k0 < q0 + 64; k0 += 32) {
    __syncthreads();  // previous iteration consumers done with Ks/Vs
#if HAVE_TDM
    if (wave == 0) {  // 32x64 K tile, row stride HKV*DH
      tdm_load_2d(Kt + (((size_t)(b * S + k0) * HKV) + hk) * DH,
                  lds_offset_of(Ks), 64u, 32u, 64u, (unsigned)(32 * HKV),
                  (unsigned)(HKV * DH));
    }
#else
    {
      int key = tid >> 2, dp = (tid & 3) * 16;
      const uint4* g = (const uint4*)(Kt +
          (((size_t)(b * S + k0 + key) * HKV) + hk) * DH + dp);
      uint4* s = (uint4*)(Ks + key * 64 + dp);
      s[0] = g[0]; s[1] = g[1];
    }
#endif
    {  // V tile transposed -> Vs[d][key] (overlaps with the K DMA)
      int key = tid >> 2, dp = (tid & 3) * 16;
      union { uint4 u[2]; __bf16 e[16]; } t;
      const uint4* g = (const uint4*)(Vt +
          (((size_t)(b * S + k0 + key) * HKV) + hk) * DH + dp);
      t.u[0] = g[0]; t.u[1] = g[1];
#pragma unroll
      for (int i = 0; i < 16; ++i) Vs[(dp + i) * 32 + key] = t.e[i];
    }
#if HAVE_TDM
    if (wave == 0) __builtin_amdgcn_s_wait_tensorcnt(0);  // Q (iter 0) + K
#endif
    __syncthreads();

    // scores: S0 = Q(16x64) K0^T, S1 = Q K1^T
    v8f s0 = vzero, s1 = vzero;
#pragma unroll
    for (int dc = 0; dc < 64; dc += 32) {
      v16bf aq = load_a_frag(Qs, wm, lane, 64, dc);
      v16bf bk0 = load_b_frag(Ks, 0, lane, 64, dc);
      s0 = __builtin_amdgcn_wmma_f32_16x16x32_bf16(false, aq, false, bk0,
                                                   (short)0, s0, false, false);
      v16bf bk1 = load_b_frag(Ks, 16, lane, 64, dc);
      s1 = __builtin_amdgcn_wmma_f32_16x16x32_bf16(false, aq, false, bk1,
                                                   (short)0, s1, false, false);
    }

    // mask + online softmax; row j+8*hl lives in vgpr j, reduce over 16 N-lanes
#pragma unroll
    for (int j = 0; j < 8; ++j) {
      int rowg = q0 + wm + j + 8 * hl;
      float v0 = s0[j] * 0.125f;
      float v1 = s1[j] * 0.125f;
      if (k0 + c > rowg) v0 = -__builtin_inff();
      if (k0 + 16 + c > rowg) v1 = -__builtin_inff();
      float rm = fmaxf(v0, v1);
      rm = fmaxf(rm, __shfl_xor(rm, 1, 32));
      rm = fmaxf(rm, __shfl_xor(rm, 2, 32));
      rm = fmaxf(rm, __shfl_xor(rm, 4, 32));
      rm = fmaxf(rm, __shfl_xor(rm, 8, 32));
      float mn = fmaxf(m_st[j], rm);
      float a_, e0, e1;
      if (mn == -__builtin_inff()) {
        a_ = 1.f; e0 = 0.f; e1 = 0.f;   // fully-masked tile for this row
      } else {
        a_ = __expf(m_st[j] - mn);
        e0 = __expf(v0 - mn);
        e1 = __expf(v1 - mn);
      }
      float rs = e0 + e1;
      rs += __shfl_xor(rs, 1, 32);
      rs += __shfl_xor(rs, 2, 32);
      rs += __shfl_xor(rs, 4, 32);
      rs += __shfl_xor(rs, 8, 32);
      l_st[j] = l_st[j] * a_ + rs;
      m_st[j] = mn;
      alpha[j] = a_; p0[j] = e0; p1[j] = e1;
    }

#pragma unroll
    for (int n = 0; n < 4; ++n)
#pragma unroll
      for (int j = 0; j < 8; ++j) acc[n][j] *= alpha[j];

    // restripe probs C-layout -> A-layout through wave-private LDS
    __bf16* pw = Ps[wave];
#pragma unroll
    for (int j = 0; j < 8; ++j) {
      pw[(j + 8 * hl) * 32 + c] = (__bf16)p0[j];
      pw[(j + 8 * hl) * 32 + 16 + c] = (__bf16)p1[j];
    }
    asm volatile("s_wait_dscnt 0" ::: "memory");  // intra-wave cross-lane LDS
    v16bf pa = load_a_frag(pw, 0, lane, 32, 0);
#pragma unroll
    for (int n = 0; n < 4; ++n) {
      v16bf bv = load_b_frag(Vs, n * 16, lane, 32, 0);
      acc[n] = __builtin_amdgcn_wmma_f32_16x16x32_bf16(false, pa, false, bv,
                                                       (short)0, acc[n], false,
                                                       false);
    }
  }

  // epilogue: O = acc / l
#pragma unroll
  for (int j = 0; j < 8; ++j) {
    float inv = (l_st[j] > 0.f) ? 1.f / l_st[j] : 0.f;
    int row = q0 + wm + j + 8 * hl;
#pragma unroll
    for (int n = 0; n < 4; ++n)
      O[(((size_t)(b * S + row) * HQ) + h) * DH + n * 16 + c] =
          (__bf16)(acc[n][j] * inv);
  }
}

// ---------------------------------------------------------------------------
extern "C" void kernel_launch(void* const* d_in, const int* in_sizes, int n_in,
                              void* d_out, int out_size, void* d_ws,
                              size_t ws_size, hipStream_t stream) {
  (void)in_sizes; (void)n_in; (void)out_size; (void)ws_size;
  const int B = 2, S = 2048, D = 2048, HQ = 32, HKV = 8, DH = 64;
  const int MS = B * S;  // 4096 rows

  const float* X  = (const float*)d_in[0];
  const float* Wq = (const float*)d_in[1];
  const float* Wk = (const float*)d_in[2];
  const float* Wv = (const float*)d_in[3];
  const float* Wo = (const float*)d_in[4];

  char* ws = (char*)d_ws;
  size_t off = 0;
  auto alloc = [&](size_t elems) {
    __bf16* p = (__bf16*)(ws + off);
    off += ((elems * sizeof(__bf16) + 255) / 256) * 256;
    return p;
  };
  __bf16* Xb  = alloc((size_t)MS * D);
  __bf16* Wqb = alloc((size_t)D * HQ * DH);
  __bf16* Wkb = alloc((size_t)D * HKV * DH);
  __bf16* Wvb = alloc((size_t)D * HKV * DH);
  __bf16* Wob = alloc((size_t)HQ * DH * D);
  __bf16* Qb  = alloc((size_t)MS * HQ * DH);
  __bf16* Kb  = alloc((size_t)MS * HKV * DH);
  __bf16* Vb  = alloc((size_t)MS * HKV * DH);
  __bf16* Ob  = alloc((size_t)MS * HQ * DH);

  cvt_bf16<<<2048, 256, 0, stream>>>(X, Xb, (size_t)MS * D);
  cvt_bf16<<<2048, 256, 0, stream>>>(Wq, Wqb, (size_t)D * HQ * DH);
  cvt_bf16<<<1024, 256, 0, stream>>>(Wk, Wkb, (size_t)D * HKV * DH);
  cvt_bf16<<<1024, 256, 0, stream>>>(Wv, Wvb, (size_t)D * HKV * DH);
  cvt_bf16<<<2048, 256, 0, stream>>>(Wo, Wob, (size_t)HQ * DH * D);

  gemm_bf16<__bf16><<<dim3((HQ * DH) / 128, MS / 64), 128, 0, stream>>>(
      Xb, Wqb, Qb, MS, HQ * DH, D);
  gemm_bf16<__bf16><<<dim3((HKV * DH) / 128, MS / 64), 128, 0, stream>>>(
      Xb, Wkb, Kb, MS, HKV * DH, D);
  gemm_bf16<__bf16><<<dim3((HKV * DH) / 128, MS / 64), 128, 0, stream>>>(
      Xb, Wvb, Vb, MS, HKV * DH, D);

  {
    int totq = B * S * HQ * 32;
    rope_kernel<<<(totq + 255) / 256, 256, 0, stream>>>(Qb, HQ, totq);
    int totk = B * S * HKV * 32;
    rope_kernel<<<(totk + 255) / 256, 256, 0, stream>>>(Kb, HKV, totk);
  }

  flash_attn<<<dim3(S / 64, HQ, B), 128, 0, stream>>>(Qb, Kb, Vb, Ob);

  gemm_bf16<float><<<dim3(D / 128, MS / 64), 128, 0, stream>>>(
      Ob, Wob, (float*)d_out, MS, D, HQ * DH);
}
